// SpikeEncoder_76914274337329
// MI455X (gfx1250) — matrix operand order repaired and checked by reading, not compiled, not run on hardware
//
#include <hip/hip_runtime.h>

#define H_    128
#define W_    128
#define PIX   (H_ * W_)        // 16384
#define NBINS 32
#define BATCH 8
#define PAD_  2
#define EPS_  1e-5f

typedef __attribute__((ext_vector_type(16))) _Float16 v16h;
typedef __attribute__((ext_vector_type(8)))  float    v8f;
typedef __attribute__((ext_vector_type(4)))  float    fv4;   // native vector for NT loads

// ---------------- Stage 0: zero histogram (B128 stores) ----------------
__global__ void se_zero_f32(float4* __restrict__ p, int n4) {
    int i = blockIdx.x * blockDim.x + threadIdx.x;
    int stride = gridDim.x * blockDim.x;
    float4 z = {0.0f, 0.0f, 0.0f, 0.0f};
    for (; i < n4; i += stride) p[i] = z;
}

// ---------------- Stage 1: event scatter ----------------
// blockIdx.y = batch (no per-event integer division).
// Events are streamed once: non-temporal loads keep the L2 free for the
// 16 MB histogram the atomics are hitting.
__global__ void se_scatter(const fv4* __restrict__ ev, int nPerBatch,
                           float* __restrict__ counts) {
    int b = blockIdx.y;
    const fv4* evb = ev + (size_t)b * nPerBatch;
    float* cb = counts + (size_t)b * NBINS * PIX;
    int i = blockIdx.x * blockDim.x + threadIdx.x;
    int stride = gridDim.x * blockDim.x;
    for (; i < nPerBatch; i += stride) {
        fv4 e = __builtin_nontemporal_load(&evb[i]);   // TH=NT stream
        int x = (int)e.x; x = x < 0 ? 0 : (x > H_ - 1 ? H_ - 1 : x);
        int y = (int)e.y; y = y < 0 ? 0 : (y > W_ - 1 ? W_ - 1 : y);
        float t = fminf(fmaxf(e.z, 0.0f), 1.0f);
        int bin = (int)(t * (float)NBINS);
        bin = bin > NBINS - 1 ? NBINS - 1 : bin;
        // no-return atomic -> global_atomic_add_f32 (STOREcnt path), L2-resident
        atomicAdd(&cb[bin * PIX + x * W_ + y], 1.0f);
    }
}

// ---------------- Stage 2: depthwise conv over bins via WMMA ----------------
// OUT[b, 0..31, p..p+15] = Mband(32x32) @ S(32 x 16), Mband[m][k] = g[k-m+2].
// One wave handles one batch x 16-pixel tile with exactly 2 v_wmma_f32_16x16x32_f16.
__global__ void se_conv_wmma(const float* __restrict__ counts,
                             const float* __restrict__ gker,
                             float* __restrict__ smoothed) {
    int gt   = blockIdx.x * blockDim.x + threadIdx.x;
    int wave = gt >> 5;
    int lane = gt & 31;
    const int tilesPerBatch = PIX / 16;
    int b  = wave / tilesPerBatch;          // compile-time-known divisor -> mul/shift
    int pt = (wave % tilesPerBatch) * 16;

    float g[5];
#pragma unroll
    for (int k = 0; k < 5; ++k) g[k] = gker[k];  // all rows of smooth_kernel identical

    int hi = (lane >> 4) & 1;   // lane half select
    int ln = lane & 15;
    int n  = pt + ln;           // pixel column for B/C/D
    const float* base = counts + (size_t)b * NBINS * PIX;

    v16h a0, a1, bm;
#pragma unroll
    for (int j = 0; j < 16; ++j) {
        // A layout: element j of lane -> K = (j/8)*16 + hi*8 + (j%8); row m = ln (+16 for tile 1)
        int K  = ((j >> 3) << 4) + (hi << 3) + (j & 7);
        int d0 = K - ln;
        int d1 = K - (ln + 16);
        a0[j] = (_Float16)((d0 >= -PAD_ && d0 <= PAD_) ? g[d0 + PAD_] : 0.0f);
        a1[j] = (_Float16)((d1 >= -PAD_ && d1 <= PAD_) ? g[d1 + PAD_] : 0.0f);
        // B layout: element j of lane -> K(bin) = (j/8)*16 + hi*8 + (j%8), column = ln
        int bin = ((j >> 3) << 4) + (hi << 3) + (j & 7);
        bm[j] = (_Float16)base[bin * PIX + n];   // counts are small ints: exact in f16
    }

    v8f c0 = {}, c1 = {};
    c0 = __builtin_amdgcn_wmma_f32_16x16x32_f16(false, a0, false, bm, (short)0, c0, false, false);
    c1 = __builtin_amdgcn_wmma_f32_16x16x32_f16(false, a1, false, bm, (short)0, c1, false, false);

    float* outb = smoothed + (size_t)b * NBINS * PIX;
    int mr = hi << 3;  // D layout: VGPR r -> row r (+8 for upper lane half)
#pragma unroll
    for (int r = 0; r < 8; ++r) {
        outb[(mr + r) * PIX + n]      = c0[r];   // out bins 0..15
        outb[(16 + mr + r) * PIX + n] = c1[r];   // out bins 16..31
    }
}

// ---------------- Stage 3: LayerNorm over pixels, per (batch, bin) ----------------
// One 256-thread block per row of 16384; float4 (B128) traffic throughout.
__global__ void se_pixel_ln(const float4* __restrict__ x, const float4* __restrict__ w,
                            const float4* __restrict__ bia, float4* __restrict__ y) {
    __shared__ float ss[256], ss2[256];
    const int n4 = PIX / 4;                    // 4096 float4 per row
    int row = blockIdx.x;                      // b*NBINS + bin
    const float4* xr = x + (size_t)row * n4;
    float s = 0.0f, s2 = 0.0f;
    for (int i = threadIdx.x; i < n4; i += 256) {
        float4 v = xr[i];
        s  += v.x + v.y + v.z + v.w;
        s2 += v.x * v.x + v.y * v.y + v.z * v.z + v.w * v.w;
    }
    ss[threadIdx.x] = s; ss2[threadIdx.x] = s2;
    __syncthreads();
    for (int off = 128; off > 0; off >>= 1) {
        if ((int)threadIdx.x < off) {
            ss[threadIdx.x]  += ss[threadIdx.x + off];
            ss2[threadIdx.x] += ss2[threadIdx.x + off];
        }
        __syncthreads();
    }
    float mu  = ss[0] * (1.0f / PIX);
    float var = ss2[0] * (1.0f / PIX) - mu * mu;
    float inv = rsqrtf(var + EPS_);
    float4* yr = y + (size_t)row * n4;
    for (int i = threadIdx.x; i < n4; i += 256) {
        float4 v = xr[i], wv = w[i], bv = bia[i], o;
        o.x = (v.x - mu) * inv * wv.x + bv.x;
        o.y = (v.y - mu) * inv * wv.y + bv.y;
        o.z = (v.z - mu) * inv * wv.z + bv.z;
        o.w = (v.w - mu) * inv * wv.w + bv.w;
        yr[i] = o;
    }
}

// ---------------- Stage 4: LayerNorm over (bins, pixels), per batch ----------------
__global__ void se_global_ln(const float4* __restrict__ x, const float4* __restrict__ w,
                             const float4* __restrict__ bia, float4* __restrict__ y) {
    __shared__ float ss[1024], ss2[1024];
    const int n  = NBINS * PIX;                // 524288
    const int n4 = n / 4;                      // 131072 float4
    int b = blockIdx.x;
    const float4* xr = x + (size_t)b * n4;
    float s = 0.0f, s2 = 0.0f;
    for (int i = threadIdx.x; i < n4; i += 1024) {
        float4 v = xr[i];
        s  += v.x + v.y + v.z + v.w;
        s2 += v.x * v.x + v.y * v.y + v.z * v.z + v.w * v.w;
    }
    ss[threadIdx.x] = s; ss2[threadIdx.x] = s2;
    __syncthreads();
    for (int off = 512; off > 0; off >>= 1) {
        if ((int)threadIdx.x < off) {
            ss[threadIdx.x]  += ss[threadIdx.x + off];
            ss2[threadIdx.x] += ss2[threadIdx.x + off];
        }
        __syncthreads();
    }
    float mu  = ss[0] * (1.0f / n);
    float var = ss2[0] * (1.0f / n) - mu * mu;
    float inv = rsqrtf(var + EPS_);
    float4* yr = y + (size_t)b * n4;
    for (int i = threadIdx.x; i < n4; i += 1024) {
        float4 v = xr[i], wv = w[i], bv = bia[i], o;
        o.x = (v.x - mu) * inv * wv.x + bv.x;
        o.y = (v.y - mu) * inv * wv.y + bv.y;
        o.z = (v.z - mu) * inv * wv.z + bv.z;
        o.w = (v.w - mu) * inv * wv.w + bv.w;
        yr[i] = o;
    }
}

extern "C" void kernel_launch(void* const* d_in, const int* in_sizes, int n_in,
                              void* d_out, int out_size, void* d_ws, size_t ws_size,
                              hipStream_t stream) {
    const float* events = (const float*)d_in[0];
    // d_in[1] = bin_edges (uniform linspace -> bin = floor(t*NBINS), boundary-consistent)
    const float* gker = (const float*)d_in[2];
    const float* pw   = (const float*)d_in[3];
    const float* pb   = (const float*)d_in[4];
    const float* gw   = (const float*)d_in[5];
    const float* gb   = (const float*)d_in[6];
    float* out = (float*)d_out;

    int totalEvents = in_sizes[0] / 4;
    int nPerBatch   = totalEvents / BATCH;
    const int nhist = BATCH * NBINS * PIX;     // 4,194,304 floats = 16 MB

    float* counts   = (float*)d_ws;                                          // 16 MB
    float* smoothed = (float*)((char*)d_ws + (size_t)nhist * sizeof(float)); // 16 MB

    se_zero_f32<<<1024, 256, 0, stream>>>((float4*)counts, nhist / 4);

    dim3 sgrid(2048, BATCH, 1);
    se_scatter<<<sgrid, 256, 0, stream>>>((const fv4*)events, nPerBatch, counts);

    // BATCH * PIX/16 waves * 32 lanes = 262144 threads -> 1024 blocks of 256
    se_conv_wmma<<<(BATCH * (PIX / 16) * 32) / 256, 256, 0, stream>>>(counts, gker, smoothed);

    // reuse counts buffer for the pixel-normed intermediate (fully overwritten)
    se_pixel_ln<<<BATCH * NBINS, 256, 0, stream>>>((const float4*)smoothed, (const float4*)pw,
                                                   (const float4*)pb, (float4*)counts);
    se_global_ln<<<BATCH, 1024, 0, stream>>>((const float4*)counts, (const float4*)gw,
                                             (const float4*)gb, (float4*)out);
}
